// AdjustableEmbeddingLM_50680614092903
// MI455X (gfx1250) — matrix operand order repaired and checked by reading, not compiled, not run on hardware
//
#include <hip/hip_runtime.h>
#include <hip/hip_bf16.h>
#include <math.h>
#include <stdint.h>

typedef __attribute__((ext_vector_type(16))) __bf16 v16bf;
typedef __attribute__((ext_vector_type(8)))  float  v8f;

constexpr int kV  = 32000;
constexpr int kE  = 512;
constexpr int kNH = 8;
constexpr int kHD = 64;   // kE / kNH
constexpr int kNL = 4;
constexpr int kFF = 2048;
constexpr int kB  = 32;
constexpr int kL  = 80;
constexpr int kM  = kB * kL;   // 2560 token rows

static __device__ __forceinline__ unsigned short bfbits(float f) {
  union { float f; unsigned u; } v; v.f = f;
  unsigned r = v.u + 0x7FFFu + ((v.u >> 16) & 1u);   // round-to-nearest-even
  return (unsigned short)(r >> 16);
}
static __device__ __forceinline__ __bf16 f2bf(float f) {
  unsigned short h = bfbits(f);
  __bf16 o; __builtin_memcpy(&o, &h, 2); return o;
}

// ---------------------------------------------------------------------------
// Bulk fp32 -> bf16 convert (weights), 4 elems per thread.
// ---------------------------------------------------------------------------
__global__ __launch_bounds__(256)
void cvt_f32_bf16(const float4* __restrict__ in, uint2* __restrict__ out, int n4)
{
  int i = blockIdx.x * blockDim.x + threadIdx.x;
  if (i >= n4) return;
  float4 f = in[i];
  uint2 o;
  o.x = (unsigned)bfbits(f.x) | ((unsigned)bfbits(f.y) << 16);
  o.y = (unsigned)bfbits(f.z) | ((unsigned)bfbits(f.w) << 16);
  out[i] = o;
}

// ---------------------------------------------------------------------------
// C[M,N] = A[M,K] @ W[N,K]^T + bias.  A, W bf16; fp32 accum via WMMA.
// Block: 256 threads (8 waves), tile BM=32 x BN=128, BK=32, double-buffered.
// Wave (wm 0..1, wn 0..3) owns a 16x32 strip: 2 WMMA / K-step sharing fa.
// Staging: gfx1250 async global->LDS copies (ASYNCcnt), one tile in flight.
// outmode: 0 = f32 store, 1 = f32 nontemporal store, 2 = bf16 store
// ---------------------------------------------------------------------------
__global__ __launch_bounds__(256)
void gemm_bf16_wmma(const __bf16* __restrict__ A, const __bf16* __restrict__ W,
                    const float* __restrict__ bias, void* __restrict__ Cv,
                    int M, int N, int K, int relu, int outmode)
{
  __shared__ __bf16 As[2 * 32 * 32];    //  4 KB (double buffer)
  __shared__ __bf16 Bs[2 * 128 * 32];   // 16 KB (double buffer)

  const int tid  = threadIdx.x;
  const int lane = tid & 31;
  const int wave = tid >> 5;
  const int wm   = wave >> 2;    // 0..1
  const int wn   = wave & 3;     // 0..3
  const int hf   = lane >> 4;    // lane half
  const int l15  = lane & 15;
  const int bn0  = blockIdx.x * 128;
  const int bm0  = blockIdx.y * 32;

  const unsigned ldsA = (unsigned)(uintptr_t)&As[0];
  const unsigned ldsB = (unsigned)(uintptr_t)&Bs[0];

  // per-thread staging coordinates (constant across K loop)
  const int a_i4  = tid * 4;                 // 4 bf16 (8B) of A tile
  const int a_row = a_i4 >> 5, a_col = a_i4 & 31;
  const int b_i8a = tid * 8;                 // 8 bf16 (16B) of B tile, part 1
  const int b_r0  = b_i8a >> 5, b_c0 = b_i8a & 31;
  const int b_i8b = (tid + 256) * 8;         // part 2
  const int b_r1  = b_i8b >> 5, b_c1 = b_i8b & 31;

  // issue one K-tile's async global->LDS copies into buffer `buf`
  auto stage = [&](int buf, int k0) {
    unsigned aofs = (unsigned)buf * (32 * 32 * 2);
    unsigned bofs = (unsigned)buf * (128 * 32 * 2);
    unsigned offA  = (unsigned)((bm0 + a_row) * K + (k0 + a_col)) * 2u;
    unsigned offB0 = (unsigned)((bn0 + b_r0) * K + (k0 + b_c0)) * 2u;
    unsigned offB1 = (unsigned)((bn0 + b_r1) * K + (k0 + b_c1)) * 2u;
    unsigned dA  = ldsA + aofs + (unsigned)a_i4 * 2u;
    unsigned dB0 = ldsB + bofs + (unsigned)b_i8a * 2u;
    unsigned dB1 = ldsB + bofs + (unsigned)b_i8b * 2u;
    asm volatile("global_load_async_to_lds_b64 %0, %1, %2"
                 :: "v"(dA), "v"(offA), "s"(A) : "memory");
    asm volatile("global_load_async_to_lds_b128 %0, %1, %2"
                 :: "v"(dB0), "v"(offB0), "s"(W) : "memory");
    asm volatile("global_load_async_to_lds_b128 %0, %1, %2"
                 :: "v"(dB1), "v"(offB1), "s"(W) : "memory");
  };

  v8f acc0 = {}, acc1 = {};
  const int nK = K >> 5;

  stage(0, 0);   // prologue: tile 0 -> buffer 0

  for (int kt = 0; kt < nK; ++kt) {
    const int cur = kt & 1;
    // my writes of tile kt have landed:
    asm volatile("s_wait_asynccnt 0x0" ::: "memory");
    // everyone's writes landed; everyone's reads of the other buffer done
    // (compiler's barrier includes s_wait_dscnt 0):
    __syncthreads();
    if (kt + 1 < nK)                       // uniform branch
      stage((kt + 1) & 1, (kt + 1) << 5);  // overlap next copy with this WMMA

    const int aB = cur * 1024;             // bf16 element offsets
    const int bB = cur * 4096;

    // A fragment (16x32 bf16): 16 contiguous K per lane (+16*hf)
    v16bf fa, fb0, fb1;
    #pragma unroll
    for (int j = 0; j < 16; ++j) {
      int ka = j + hf * 16;
      fa[j] = As[aB + (wm * 16 + l15) * 32 + ka];
    }
    // B fragments (32x16): lane = n, 16 contiguous K per lane (+16*hf)
    #pragma unroll
    for (int j = 0; j < 16; ++j) {
      fb0[j] = Bs[bB + (wn * 32 + l15) * 32 + (j + 16 * hf)];
      fb1[j] = Bs[bB + (wn * 32 + 16 + l15) * 32 + (j + 16 * hf)];
    }
    acc0 = __builtin_amdgcn_wmma_f32_16x16x32_bf16(
        false, fa, false, fb0, (short)0, acc0, false, false);
    acc1 = __builtin_amdgcn_wmma_f32_16x16x32_bf16(
        false, fa, false, fb1, (short)0, acc1, false, false);
  }

  const int n0 = bn0 + wn * 32 + l15;
  const int n1 = n0 + 16;
  const float bv0 = bias ? bias[n0] : 0.0f;
  const float bv1 = bias ? bias[n1] : 0.0f;
  #pragma unroll
  for (int r = 0; r < 8; ++r) {
    int m = bm0 + wm * 16 + r + hf * 8;   // C/D layout: VGPR r -> row r (+8 hi half)
    float v0 = acc0[r] + bv0;
    float v1 = acc1[r] + bv1;
    if (relu) { v0 = fmaxf(v0, 0.0f); v1 = fmaxf(v1, 0.0f); }
    size_t o0 = (size_t)m * N + n0, o1 = (size_t)m * N + n1;
    if (outmode == 2) {
      ((__bf16*)Cv)[o0] = f2bf(v0);
      ((__bf16*)Cv)[o1] = f2bf(v1);
    } else if (outmode == 1) {
      __builtin_nontemporal_store(v0, &((float*)Cv)[o0]);
      __builtin_nontemporal_store(v1, &((float*)Cv)[o1]);
    } else {
      ((float*)Cv)[o0] = v0;
      ((float*)Cv)[o1] = v1;
    }
  }
}

// ---------------------------------------------------------------------------
// x[b,l,:] = W_in[:, caps[b,l]] + b_in + pos_emb[l]   (one-hot matmul == gather)
// writes fp32 master + bf16 shadow
// ---------------------------------------------------------------------------
__global__ void embed_kernel(const int* __restrict__ caps,
                             const float* __restrict__ W_in,
                             const float* __restrict__ b_in,
                             const float* __restrict__ pos,
                             float* __restrict__ X, __bf16* __restrict__ Xbf)
{
  int idx = blockIdx.x * blockDim.x + threadIdx.x;
  if (idx >= kM * kE) return;
  int bl = idx >> 9;        // / 512
  int e  = idx & 511;
  int l  = bl % kL;
  int tok = caps[bl];
  float v = W_in[(size_t)e * kV + tok] + b_in[e] + pos[l * kE + e];
  X[idx] = v;
  Xbf[idx] = f2bf(v);
}

// ---------------------------------------------------------------------------
// Self-attention, one block per (b, h, q). Causal softmax over keys <= q.
// Output written directly as bf16 (next consumer is a bf16 GEMM).
// ---------------------------------------------------------------------------
__global__ __launch_bounds__(128)
void attn_kernel(const float* __restrict__ QKV, __bf16* __restrict__ O)
{
  int gid = blockIdx.x;
  int q  = gid % kL;
  int bh = gid / kL;
  int h  = bh % kNH;
  int b  = bh / kNH;
  int tid = threadIdx.x;

  __shared__ float qv[kHD];
  __shared__ float sc[kL];
  __shared__ float red[2];

  const size_t rowstride = 3 * kE;
  const float* qptr = QKV + (size_t)(b * kL + q) * rowstride + h * kHD;
  if (tid < kHD) qv[tid] = qptr[tid];
  __syncthreads();

  const float scale = 0.125f;   // 1/sqrt(64)
  for (int k = tid; k <= q; k += 128) {
    const float* kptr = QKV + (size_t)(b * kL + k) * rowstride + kE + h * kHD;
    float s = 0.f;
    #pragma unroll 8
    for (int d = 0; d < kHD; ++d) s += qv[d] * kptr[d];
    sc[k] = s * scale;
  }
  __syncthreads();
  if (tid == 0) {
    float mx = -1e30f;
    for (int k = 0; k <= q; ++k) mx = fmaxf(mx, sc[k]);
    red[0] = mx;
  }
  __syncthreads();
  float mx = red[0];
  for (int k = tid; k <= q; k += 128) sc[k] = __expf(sc[k] - mx);
  __syncthreads();
  if (tid == 0) {
    float sm = 0.f;
    for (int k = 0; k <= q; ++k) sm += sc[k];
    red[1] = 1.0f / sm;
  }
  __syncthreads();
  float inv = red[1];
  if (tid < kHD) {
    float o = 0.f;
    for (int k = 0; k <= q; ++k) {
      const float* vptr = QKV + (size_t)(b * kL + k) * rowstride + 2 * kE + h * kHD;
      o += sc[k] * vptr[tid];
    }
    O[(size_t)(b * kL + q) * kE + h * kHD + tid] = f2bf(o * inv);
  }
}

// ---------------------------------------------------------------------------
// Cross-attn with zero memory collapses to a constant vector:
//   ca_vec = ca_v_bias @ ca_out_w^T + ca_out_b     (softmax over 1 key == 1)
// ---------------------------------------------------------------------------
__global__ void cavec_kernel(const float* __restrict__ ca_out_w,
                             const float* __restrict__ ca_out_b,
                             const float* __restrict__ vbias,
                             float* __restrict__ out)
{
  int e = blockIdx.x * blockDim.x + threadIdx.x;
  if (e >= kE) return;
  float s = ca_out_b[e];
  const float* wrow = ca_out_w + (size_t)e * kE;
  for (int f = 0; f < kE; ++f) s += wrow[f] * vbias[f];
  out[e] = s;
}

// ---------------------------------------------------------------------------
// In-place X = LayerNorm(X + Y), Y per-row or broadcast. One block per row.
// Writes fp32 master + bf16 shadow for the following GEMM.
// ---------------------------------------------------------------------------
__global__ __launch_bounds__(256)
void add_ln_kernel(float* __restrict__ X, __bf16* __restrict__ Xbf,
                   const float* __restrict__ Y,
                   const float* __restrict__ w, const float* __restrict__ bp,
                   int ybroadcast)
{
  int row = blockIdx.x;
  int tid = threadIdx.x;
  __shared__ float r1[256], r2[256];

  size_t base = (size_t)row * kE;
  float y0 = ybroadcast ? Y[tid]       : Y[base + tid];
  float y1 = ybroadcast ? Y[tid + 256] : Y[base + tid + 256];
  float v0 = X[base + tid] + y0;
  float v1 = X[base + tid + 256] + y1;

  r1[tid] = v0 + v1;
  r2[tid] = v0 * v0 + v1 * v1;
  __syncthreads();
  for (int s = 128; s > 0; s >>= 1) {
    if (tid < s) { r1[tid] += r1[tid + s]; r2[tid] += r2[tid + s]; }
    __syncthreads();
  }
  float mean = r1[0] * (1.0f / kE);
  float var  = r2[0] * (1.0f / kE) - mean * mean;
  float rstd = rsqrtf(var + 1e-5f);
  float o0 = (v0 - mean) * rstd * w[tid]       + bp[tid];
  float o1 = (v1 - mean) * rstd * w[tid + 256] + bp[tid + 256];
  X[base + tid]         = o0;
  X[base + tid + 256]   = o1;
  Xbf[base + tid]       = f2bf(o0);
  Xbf[base + tid + 256] = f2bf(o1);
}

// ---------------------------------------------------------------------------
static inline void cvt(const float* in, __bf16* out, size_t n, hipStream_t s) {
  int n4 = (int)(n / 4);
  cvt_f32_bf16<<<(n4 + 255) / 256, 256, 0, s>>>((const float4*)in, (uint2*)out, n4);
}

extern "C" void kernel_launch(void* const* d_in, const int* in_sizes, int n_in,
                              void* d_out, int out_size, void* d_ws, size_t ws_size,
                              hipStream_t stream)
{
  (void)in_sizes; (void)n_in; (void)out_size; (void)ws_size;

  const int*   caps     = (const int*)  d_in[0];
  const float* W_in     = (const float*)d_in[1];
  const float* b_in     = (const float*)d_in[2];
  const float* pos      = (const float*)d_in[3];
  const float* sa_in_w  = (const float*)d_in[4];
  const float* sa_in_b  = (const float*)d_in[5];
  const float* sa_out_w = (const float*)d_in[6];
  const float* sa_out_b = (const float*)d_in[7];
  /* ca_in_w (d_in[8]) dead: mem==0 and softmax over a single key == 1 */
  const float* ca_in_b  = (const float*)d_in[9];
  const float* ca_out_w = (const float*)d_in[10];
  const float* ca_out_b = (const float*)d_in[11];
  const float* ff1_w    = (const float*)d_in[12];
  const float* ff1_b    = (const float*)d_in[13];
  const float* ff2_w    = (const float*)d_in[14];
  const float* ff2_b    = (const float*)d_in[15];
  const float* ln1_w    = (const float*)d_in[16];
  const float* ln1_b    = (const float*)d_in[17];
  const float* ln2_w    = (const float*)d_in[18];
  const float* ln2_b    = (const float*)d_in[19];
  const float* ln3_w    = (const float*)d_in[20];
  const float* ln3_b    = (const float*)d_in[21];
  const float* out_w    = (const float*)d_in[22];
  const float* out_b    = (const float*)d_in[23];

  // ---- workspace layout (~100 MB) ----
  char* p = (char*)d_ws;
  float* X    = (float*)p;  p += (size_t)kM * kE * 4;          // fp32 residual stream
  float* QKV  = (float*)p;  p += (size_t)kM * 3 * kE * 4;      // fp32 qkv
  float* TMP  = (float*)p;  p += (size_t)kM * kE * 4;          // fp32 sublayer out
  float* CAV  = (float*)p;  p += (size_t)kE * 4;               // cross-attn const vec
  __bf16* Xbf    = (__bf16*)p; p += (size_t)kM * kE * 2;       // bf16 shadows
  __bf16* ATTNbf = (__bf16*)p; p += (size_t)kM * kE * 2;
  __bf16* Hbf    = (__bf16*)p; p += (size_t)kM * kFF * 2;
  __bf16* w_sain  = (__bf16*)p; p += (size_t)kNL * 3 * kE * kE * 2;  // bf16 weights
  __bf16* w_saout = (__bf16*)p; p += (size_t)kNL * kE * kE * 2;
  __bf16* w_ff1   = (__bf16*)p; p += (size_t)kNL * kFF * kE * 2;
  __bf16* w_ff2   = (__bf16*)p; p += (size_t)kNL * kE * kFF * 2;
  __bf16* w_out   = (__bf16*)p; p += (size_t)kV * kE * 2;

  // one-time (per launch) weight conversion to bf16
  cvt(sa_in_w,  w_sain,  (size_t)kNL * 3 * kE * kE, stream);
  cvt(sa_out_w, w_saout, (size_t)kNL * kE * kE,     stream);
  cvt(ff1_w,    w_ff1,   (size_t)kNL * kFF * kE,    stream);
  cvt(ff2_w,    w_ff2,   (size_t)kNL * kE * kFF,    stream);
  cvt(out_w,    w_out,   (size_t)kV * kE,           stream);

  embed_kernel<<<(kM * kE + 255) / 256, 256, 0, stream>>>(caps, W_in, b_in, pos, X, Xbf);

  for (int i = 0; i < kNL; ++i) {
    // self-attention
    gemm_bf16_wmma<<<dim3((3 * kE) / 128, kM / 32), 256, 0, stream>>>(
        Xbf, w_sain + (size_t)i * 3 * kE * kE, sa_in_b + (size_t)i * 3 * kE,
        QKV, kM, 3 * kE, kE, 0, 0);
    attn_kernel<<<kB * kNH * kL, 128, 0, stream>>>(QKV, ATTNbf);
    gemm_bf16_wmma<<<dim3(kE / 128, kM / 32), 256, 0, stream>>>(
        ATTNbf, w_saout + (size_t)i * kE * kE, sa_out_b + (size_t)i * kE,
        TMP, kM, kE, kE, 0, 0);
    add_ln_kernel<<<kM, 256, 0, stream>>>(X, Xbf, TMP, ln1_w + i * kE, ln1_b + i * kE, 0);

    // cross-attention (constant vector) + LN2
    cavec_kernel<<<2, 256, 0, stream>>>(
        ca_out_w + (size_t)i * kE * kE, ca_out_b + (size_t)i * kE,
        ca_in_b + (size_t)i * 3 * kE + 2 * kE, CAV);
    add_ln_kernel<<<kM, 256, 0, stream>>>(X, Xbf, CAV, ln2_w + i * kE, ln2_b + i * kE, 1);

    // FFN
    gemm_bf16_wmma<<<dim3(kFF / 128, kM / 32), 256, 0, stream>>>(
        Xbf, w_ff1 + (size_t)i * kFF * kE, ff1_b + (size_t)i * kFF,
        Hbf, kM, kFF, kE, 1, 2);
    gemm_bf16_wmma<<<dim3(kE / 128, kM / 32), 256, 0, stream>>>(
        Hbf, w_ff2 + (size_t)i * kE * kFF, ff2_b + (size_t)i * kE,
        TMP, kM, kE, kFF, 0, 0);
    add_ln_kernel<<<kM, 256, 0, stream>>>(X, Xbf, TMP, ln3_w + i * kE, ln3_b + i * kE, 0);
  }

  // vocab projection: compute-dominant GEMM, stream 327 MB output nontemporally
  gemm_bf16_wmma<<<dim3(kV / 128, kM / 32), 256, 0, stream>>>(
      Xbf, w_out, out_b, d_out, kM, kV, kE, 0, 1);
}
